// QMixer_81466939671033
// MI455X (gfx1250) — matrix operand order repaired and checked by reading, not compiled, not run on hardware
//
#include <hip/hip_runtime.h>
#include <hip/hip_bf16.h>

#define Bn      16384
#define NAG     32
#define EMB     32
#define HIDN    64
#define NACT    5
#define SDIM    512
#define ODIM    256

typedef __attribute__((ext_vector_type(16))) __bf16          bf16x16;
typedef __attribute__((ext_vector_type(8)))  __bf16          bf16x8;
typedef __attribute__((ext_vector_type(4)))  __bf16          bf16x4;
typedef __attribute__((ext_vector_type(16))) unsigned short  u16x16;
typedef __attribute__((ext_vector_type(8)))  unsigned short  u16x8;
typedef __attribute__((ext_vector_type(4)))  unsigned short  u16x4;
typedef __attribute__((ext_vector_type(8)))  float           f32x8;
typedef __attribute__((ext_vector_type(4)))  float           f32x4;

static __device__ __forceinline__ unsigned short f2bf(float f) {
    unsigned int u = __builtin_bit_cast(unsigned int, f);
    unsigned int r = u + 0x7FFFu + ((u >> 16) & 1u);   // round-to-nearest-even
    return (unsigned short)(r >> 16);
}

static __device__ __forceinline__ f32x8 wmma_bf16(bf16x16 a, bf16x16 b, f32x8 c) {
    return __builtin_amdgcn_wmma_f32_16x16x32_bf16(false, a, false, b, (short)0, c, false, false);
}

#define SHUF16(LO, HI) __builtin_shufflevector((LO), (HI), 0, 1, 2, 3, 4, 5, 6, 7, \
                                               8, 9, 10, 11, 12, 13, 14, 15)

// ---------------------------------------------------------------------------
// Pack a row-major f32 weight [Ktot x Ncols] into bf16, pre-swizzled into the
// CDNA5 WMMA 16x16x32 B-matrix lane layout:
//   packed[((tile*chunks + chunk)*32 + lane)*16 + h]
//   K = chunk*32 + (lane<16 ? 0:8) + (h&7) + (h>=8 ? 16:0) ; col = tile*16 + lane%16
// Hot loops then read one contiguous 32B vector per lane per (tile,chunk).
// ---------------------------------------------------------------------------
__global__ __launch_bounds__(256) void pack_b_kernel(const float* __restrict__ W,
                                                     unsigned short* __restrict__ dst,
                                                     int Ktot, int Ncols)
{
    int idx = blockIdx.x * 256 + threadIdx.x;
    int chunks = Ktot >> 5;
    int total  = (Ncols >> 4) * chunks * 512;
    if (idx >= total) return;
    int h    = idx & 15;
    int lane = (idx >> 4) & 31;
    int tc   = idx >> 9;
    int t = tc / chunks, c = tc % chunks;
    int K   = c * 32 + ((lane < 16) ? 0 : 8) + (h & 7) + ((h & 8) ? 16 : 0);
    int col = t * 16 + (lane & 15);
    dst[idx] = f2bf(W[(size_t)K * Ncols + col]);
}

// ---------------------------------------------------------------------------
// Agent net: per wave, 16 rows of flattened [B*N, 256] obs.
//   h = relu(obs @ fc1 + b)  via WMMA bf16 (8 K-chunks x 4 N-tiles)
//   qs = softmax(h @ fc2 + b), best = max(qs) = 1/sum(exp(logit-max))
// A-fragment per lane = two contiguous 8-float runs -> 2x f32x8 load,
// packed bf16 convert, one shuffle.
// ---------------------------------------------------------------------------
__global__ __launch_bounds__(128) void agents_kernel(const float* __restrict__ obs,
                                                     const unsigned short* __restrict__ fc1p,
                                                     const float* __restrict__ fc1_b,
                                                     const float* __restrict__ fc2_w,
                                                     const float* __restrict__ fc2_b,
                                                     float* __restrict__ out_qs,
                                                     float* __restrict__ best_qs)
{
    __shared__ float h_lds[4][16][HIDN + 1];
    const int wave = threadIdx.x >> 5;
    const int lane = threadIdx.x & 31;
    const size_t tile = (size_t)blockIdx.x * 4 + wave;      // 16-row tile
    const size_t row0 = tile * 16;

    const int koff = (lane < 16) ? 0 : 8;                   // A-matrix lane K offset
    const int mrow = lane & 15;
    const float* arow = obs + (row0 + mrow) * ODIM;
    __builtin_prefetch(arow, 0, 3);                         // global_prefetch_b8

    f32x8 acc[4] = {};
    for (int c = 0; c < 8; ++c) {
        const float* ap = arow + c * 32 + koff;             // 32B aligned
        f32x8 lo = *(const f32x8*)ap;                       // K = koff .. koff+7
        f32x8 hi = *(const f32x8*)(ap + 16);                // K = koff+16 .. koff+23
        bf16x8 blo = __builtin_convertvector(lo, bf16x8);
        bf16x8 bhi = __builtin_convertvector(hi, bf16x8);
        bf16x16 a = SHUF16(blo, bhi);
        #pragma unroll
        for (int t = 0; t < 4; ++t) {
            u16x16 braw = *((const u16x16*)fc1p + ((t * 8 + c) * 32 + lane));
            acc[t] = wmma_bf16(a, __builtin_bit_cast(bf16x16, braw), acc[t]);
        }
    }
    // bias + relu -> LDS  (C layout: vgpr j -> row j + 8*(lane/16), col = lane%16)
    const int crow = (lane >> 4) * 8;
    #pragma unroll
    for (int t = 0; t < 4; ++t) {
        int col = t * 16 + (lane & 15);
        float bias = fc1_b[col];
        #pragma unroll
        for (int j = 0; j < 8; ++j)
            h_lds[wave][crow + j][col] = fmaxf(acc[t][j] + bias, 0.f);
    }
    __syncthreads();

    if (lane < 16) {
        const int M = lane;
        const size_t rg = row0 + M;
        float lg[NACT];
        #pragma unroll
        for (int aI = 0; aI < NACT; ++aI) lg[aI] = fc2_b[aI];
        for (int k = 0; k < HIDN; ++k) {
            float hv = h_lds[wave][M][k];
            #pragma unroll
            for (int aI = 0; aI < NACT; ++aI) lg[aI] = fmaf(hv, fc2_w[k * NACT + aI], lg[aI]);
        }
        float m = lg[0];
        #pragma unroll
        for (int aI = 1; aI < NACT; ++aI) m = fmaxf(m, lg[aI]);
        float e[NACT], s = 0.f;
        #pragma unroll
        for (int aI = 0; aI < NACT; ++aI) { e[aI] = __expf(lg[aI] - m); s += e[aI]; }
        float inv = 1.f / s;
        #pragma unroll
        for (int aI = 0; aI < NACT; ++aI) out_qs[rg * NACT + aI] = e[aI] * inv;
        best_qs[rg] = inv;                                   // max softmax prob
    }
}

// ---------------------------------------------------------------------------
// Fused hypernet + mixer: 1 block = 16 samples, 8 waves.
//   waves 0-7: 8 col-tiles each of w1 = |states@hw1+b| folded against best_qs
//              into per-wave partial y slices (deterministic reduction)
//   wave 0: b1, wave 1: |wf|, wave 2: relu(v1 hidden)   (all via WMMA)
//   lanes 0-15: q_tot = sum(relu(y+b1)*wf) + v1h@v2 + v2_b
// ---------------------------------------------------------------------------
__global__ __launch_bounds__(256) void mixer_kernel(const float* __restrict__ states,
                                                    const unsigned short* __restrict__ hw1p,
                                                    const float* __restrict__ hw1_b,
                                                    const unsigned short* __restrict__ hb1p,
                                                    const float* __restrict__ hb1_b,
                                                    const unsigned short* __restrict__ hwfp,
                                                    const float* __restrict__ hwf_b,
                                                    const unsigned short* __restrict__ v1p,
                                                    const float* __restrict__ v1_b,
                                                    const float* __restrict__ v2_w,
                                                    const float* __restrict__ v2_b,
                                                    const float* __restrict__ best_qs,
                                                    float* __restrict__ q_tot)
{
    __shared__ unsigned short sh_s[16][SDIM + 8];           // row pad = 16B
    __shared__ float y_part[8][16][EMB];
    __shared__ float q_lds[16][NAG];
    __shared__ float b1_lds[16][EMB];
    __shared__ float wf_lds[16][EMB];
    __shared__ float vh_lds[16][EMB];

    const int tid = threadIdx.x;
    const int wave = tid >> 5, lane = tid & 31;
    const size_t row0 = (size_t)blockIdx.x * 16;

    // stage states f32 -> bf16 LDS, 4 elems per thread-iter (vector cvt)
    for (int i = tid; i < 16 * SDIM / 4; i += 256) {
        int i4 = i * 4;
        int r = i4 >> 9, k = i4 & (SDIM - 1);
        f32x4 f = *(const f32x4*)&states[(row0 + r) * SDIM + k];
        bf16x4 b = __builtin_convertvector(f, bf16x4);
        *(u16x4*)&sh_s[r][k] = __builtin_bit_cast(u16x4, b);
    }
    for (int i = tid; i < 16 * NAG; i += 256) {
        q_lds[i >> 5][i & 31] = best_qs[row0 * NAG + i];
    }
    __syncthreads();

    const int koff = (lane < 16) ? 0 : 8;
    const int M = lane & 15;
    const int crow = (lane >> 4) * 8;

    // ---- main: 8 col-tiles of hw1 per wave, K = 512 (16 chunks) ----
    f32x8 acc[8] = {};
    for (int c = 0; c < 16; ++c) {
        const unsigned short* sp = &sh_s[M][c * 32 + koff];
        u16x8 lo = *(const u16x8*)sp;                       // ds_load_b128
        u16x8 hi = *(const u16x8*)(sp + 16);                // ds_load_b128
        bf16x16 a = __builtin_bit_cast(bf16x16, SHUF16(lo, hi));
        #pragma unroll
        for (int tt = 0; tt < 8; ++tt) {
            int t = wave * 8 + tt;
            u16x16 braw = *((const u16x16*)hw1p + ((t * 16 + c) * 32 + lane));
            acc[tt] = wmma_bf16(a, __builtin_bit_cast(bf16x16, braw), acc[tt]);
        }
    }
    // fold |acc + bias| against best_qs into per-wave y partials
    {
        float yacc[2][8] = {};
        #pragma unroll
        for (int tt = 0; tt < 8; ++tt) {
            int t = wave * 8 + tt;
            int n = t >> 1;                  // agent index (tile pair)
            int par = tt & 1;                // e-half: 0 -> e 0..15, 1 -> e 16..31
            int col = t * 16 + (lane & 15);  // col into [0,1024)
            float wb = hw1_b[col];
            #pragma unroll
            for (int j = 0; j < 8; ++j) {
                float w1v = fabsf(acc[tt][j] + wb);
                yacc[par][j] += q_lds[crow + j][n] * w1v;
            }
        }
        #pragma unroll
        for (int par = 0; par < 2; ++par) {
            int e = par * 16 + (lane & 15);
            #pragma unroll
            for (int j = 0; j < 8; ++j)
                y_part[wave][crow + j][e] = yacc[par][j];
        }
    }

    // ---- side GEMMs: b1 (wave0), wf (wave1), v1 hidden (wave2), 32 cols each ----
    if (wave < 3) {
        const unsigned short* wp = (wave == 0) ? hb1p : (wave == 1) ? hwfp : v1p;
        f32x8 eacc[2] = {};
        for (int c = 0; c < 16; ++c) {
            const unsigned short* sp = &sh_s[M][c * 32 + koff];
            u16x8 lo = *(const u16x8*)sp;
            u16x8 hi = *(const u16x8*)(sp + 16);
            bf16x16 a = __builtin_bit_cast(bf16x16, SHUF16(lo, hi));
            #pragma unroll
            for (int tt = 0; tt < 2; ++tt) {
                u16x16 braw = *((const u16x16*)wp + ((tt * 16 + c) * 32 + lane));
                eacc[tt] = wmma_bf16(a, __builtin_bit_cast(bf16x16, braw), eacc[tt]);
            }
        }
        #pragma unroll
        for (int tt = 0; tt < 2; ++tt) {
            int col = tt * 16 + (lane & 15);
            float bias = (wave == 0) ? hb1_b[col] : (wave == 1) ? hwf_b[col] : v1_b[col];
            #pragma unroll
            for (int j = 0; j < 8; ++j) {
                int r = crow + j;
                float val = eacc[tt][j] + bias;
                if (wave == 0)      b1_lds[r][col] = val;
                else if (wave == 1) wf_lds[r][col] = fabsf(val);
                else                vh_lds[r][col] = fmaxf(val, 0.f);
            }
        }
    }
    __syncthreads();

    // ---- final per-sample reduction (fixed order -> deterministic) ----
    if (tid < 16) {
        int r = tid;
        float qt = 0.f;
        #pragma unroll 4
        for (int e = 0; e < EMB; ++e) {
            float y = b1_lds[r][e];
            #pragma unroll
            for (int w = 0; w < 8; ++w) y += y_part[w][r][e];
            qt = fmaf(fmaxf(y, 0.f), wf_lds[r][e], qt);
        }
        float vv = v2_b[0];
        #pragma unroll 4
        for (int e = 0; e < EMB; ++e) vv = fmaf(vh_lds[r][e], v2_w[e], vv);
        q_tot[row0 + r] = qt + vv;
    }
}

// ---------------------------------------------------------------------------
extern "C" void kernel_launch(void* const* d_in, const int* in_sizes, int n_in,
                              void* d_out, int out_size, void* d_ws, size_t ws_size,
                              hipStream_t stream)
{
    const float* states = (const float*)d_in[0];
    const float* obs    = (const float*)d_in[1];
    const float* fc1_w  = (const float*)d_in[2];
    const float* fc1_b  = (const float*)d_in[3];
    const float* fc2_w  = (const float*)d_in[4];
    const float* fc2_b  = (const float*)d_in[5];
    const float* hw1_w  = (const float*)d_in[6];
    const float* hw1_b  = (const float*)d_in[7];
    const float* hb1_w  = (const float*)d_in[8];
    const float* hb1_b  = (const float*)d_in[9];
    const float* hwf_w  = (const float*)d_in[10];
    const float* hwf_b  = (const float*)d_in[11];
    const float* v1_w   = (const float*)d_in[12];
    const float* v1_b   = (const float*)d_in[13];
    const float* v2_w   = (const float*)d_in[14];
    const float* v2_b   = (const float*)d_in[15];

    float* out_qs = (float*)d_out;                           // [B*N*5]
    float* qtot   = (float*)d_out + (size_t)Bn * NAG * NACT; // [B]

    char* ws = (char*)d_ws;
    float* best           = (float*)ws;                      // B*N floats = 2 MB
    unsigned short* fc1p  = (unsigned short*)(ws + ((size_t)Bn * NAG * 4));
    unsigned short* hw1p  = fc1p + 4 * 8 * 512;              // 64 tiles * 16 chunks
    unsigned short* hb1p  = hw1p + 64 * 16 * 512;
    unsigned short* hwfp  = hb1p + 2 * 16 * 512;
    unsigned short* v1p   = hwfp + 2 * 16 * 512;

    // pack all B-side weights to bf16 in WMMA lane layout
    pack_b_kernel<<<(4 * 8 * 512 + 255) / 256, 256, 0, stream>>>(fc1_w, fc1p, ODIM, HIDN);
    pack_b_kernel<<<(64 * 16 * 512 + 255) / 256, 256, 0, stream>>>(hw1_w, hw1p, SDIM, NAG * EMB);
    pack_b_kernel<<<(2 * 16 * 512 + 255) / 256, 256, 0, stream>>>(hb1_w, hb1p, SDIM, EMB);
    pack_b_kernel<<<(2 * 16 * 512 + 255) / 256, 256, 0, stream>>>(hwf_w, hwfp, SDIM, EMB);
    pack_b_kernel<<<(2 * 16 * 512 + 255) / 256, 256, 0, stream>>>(v1_w, v1p, SDIM, EMB);

    // agent MLP + softmax: B*N/16 = 32768 wave-tiles, 4 waves/block
    agents_kernel<<<(Bn * NAG / 16) / 4, 128, 0, stream>>>(obs, fc1p, fc1_b, fc2_w, fc2_b,
                                                           out_qs, best);

    // fused hypernet + mixing: 16 samples per block
    mixer_kernel<<<Bn / 16, 256, 0, stream>>>(states, hw1p, hw1_b, hb1p, hb1_b,
                                              hwfp, hwf_b, v1p, v1_b, v2_w, v2_b,
                                              best, qtot);
}
